// ProsodyPredictor_66812511257074
// MI455X (gfx1250) — compile-verified, matrix operand order checked by reading, not run on hardware
//
#include <hip/hip_runtime.h>
#include <hip/hip_bf16.h>
#include <stdint.h>

// ---------------------------------------------------------------------------
// ProsodyPredictor on gfx1250 (MI455X): f16 WMMA GEMMs everywhere, persistent
// LDS-resident GRU scan, async-to-LDS A-panel staging, fused epilogues.
// ---------------------------------------------------------------------------

#define ZONEOUT_F 0.1f
constexpr int Bb   = 64;
constexpr int Tt   = 1024;
constexpr int Cc   = 512;   // = D_IN = C = H
constexpr int Hh   = 512;
constexpr int G3H  = 1536;  // 3*H
constexpr int DOUT = 4;
constexpr int Gg   = 10;
constexpr int KTAP = 5;

typedef __attribute__((ext_vector_type(16))) _Float16 v16h;
typedef __attribute__((ext_vector_type(8)))  _Float16 v8h;
typedef __attribute__((ext_vector_type(8)))  float    v8f;

// ---- CDNA5 WMMA f16 A-fragment loader -------------------------------------
// ISA 16-bit A 16x32 layout: lane(0..15) row m=lane, elems j0..7 = K kh..kh+7,
// j8..15 = K kh+16..kh+23 with kh = (lane>=16 ? 8 : 0). Two contiguous 16B
// segments per lane -> 2x ds_load_b128 from an LDS tile.
static __device__ __forceinline__ v16h load_a_frag(const _Float16* base,
                                                   int rowStride, int lane) {
  const int m  = lane & 15;
  const int kh = (lane >> 4) << 3;
  const _Float16* p = base + m * rowStride + kh;
  v8h lo = *(const v8h*)(p);
  v8h hi = *(const v8h*)(p + 16);
  return __builtin_shufflevector(lo, hi, 0,1,2,3,4,5,6,7,
                                         8,9,10,11,12,13,14,15);
}

static __device__ __forceinline__ v8f wmma_f16(v16h a, v16h b, v8f c) {
  return __builtin_amdgcn_wmma_f32_16x16x32_f16(false, a, false, b,
                                                (short)0, c, false, false);
}

static __device__ __forceinline__ float sigm(float x) {
  return 1.0f / (1.0f + __expf(-x));
}

// ---- CDNA5 async copy: 16 bytes global -> LDS, tracked by ASYNCcnt ---------
// LDS operand is the wave-relative LDS byte address = low 32 bits of the
// generic shared-memory pointer.
static __device__ __forceinline__ void async_copy_b128(void* lds_dst,
                                                       const void* gsrc) {
  unsigned lds_off = (unsigned)(uintptr_t)lds_dst;
  unsigned long long ga = (unsigned long long)(uintptr_t)gsrc;
  asm volatile("global_load_async_to_lds_b128 %0, %1, off"
               :: "v"(lds_off), "v"(ga) : "memory");
}
static __device__ __forceinline__ void wait_async0() {
  asm volatile("s_wait_asynccnt 0x0" ::: "memory");
}

// ---- elementwise f32 -> f16 ------------------------------------------------
__global__ void cvt_f16_kernel(const float* __restrict__ src,
                               _Float16* __restrict__ dst, size_t n) {
  size_t i = (size_t)blockIdx.x * blockDim.x + threadIdx.x;
  if (i < n) dst[i] = (_Float16)src[i];
}

// ---- copy target_prosody to outs slot of d_out -----------------------------
__global__ void copy_outs_kernel(const float* __restrict__ src,
                                 float* __restrict__ dst, size_t n) {
  size_t i = (size_t)blockIdx.x * blockDim.x + threadIdx.x;
  if (i < n) dst[i] = src[i];
}

// ---- weight packers: f32 -> f16 in per-lane B-fragment layout --------------
// dst[((nt*chunks + ch)*32 + lane)*16 + j], n = nt*16 + (lane&15),
// kk = ch*32 + (lane>>4)*8 + (j<8 ? j : j+8)
__global__ void pack_linear_kernel(const float* __restrict__ src, int srcStride,
                                   int N, int chunks, _Float16* __restrict__ dst,
                                   size_t total) {
  size_t idx = (size_t)blockIdx.x * blockDim.x + threadIdx.x;
  if (idx >= total) return;
  int    j    = (int)(idx & 15);
  int    lane = (int)((idx >> 4) & 31);
  size_t rest = idx >> 9;
  int    ch   = (int)(rest % (size_t)chunks);
  int    nt   = (int)(rest / (size_t)chunks);
  int n  = (nt << 4) + (lane & 15);
  int kk = (ch << 5) + ((lane >> 4) << 3) + (j < 8 ? j : j + 8);
  float v = (n < N) ? src[(size_t)n * srcStride + kk] : 0.0f;
  dst[idx] = (_Float16)v;
}

// conv weight (C, Cin, 5); K-index kk = ktap*Cin + cin
__global__ void pack_conv_kernel(const float* __restrict__ src, int Cin,
                                 _Float16* __restrict__ dst, size_t total) {
  size_t idx = (size_t)blockIdx.x * blockDim.x + threadIdx.x;
  if (idx >= total) return;
  const int chunks = (Cin * KTAP) / 32;
  int    j    = (int)(idx & 15);
  int    lane = (int)((idx >> 4) & 31);
  size_t rest = idx >> 9;
  int    ch   = (int)(rest % (size_t)chunks);
  int    nt   = (int)(rest / (size_t)chunks);
  int n  = (nt << 4) + (lane & 15);
  int kk = (ch << 5) + ((lane >> 4) << 3) + (j < 8 ? j : j + 8);
  int ktap = kk / Cin, cin = kk % Cin;
  dst[idx] = (_Float16)src[((size_t)n * Cin + cin) * KTAP + ktap];
}

// combined heads weight: rows = [pi(10) | sigma(40) | mu(40)], K=1024, Npad=96
__global__ void pack_heads_kernel(const float* __restrict__ piw,
                                  const float* __restrict__ sigw,
                                  const float* __restrict__ muw,
                                  _Float16* __restrict__ dst, size_t total) {
  size_t idx = (size_t)blockIdx.x * blockDim.x + threadIdx.x;
  if (idx >= total) return;
  const int chunks = 32;  // K = 1024
  int    j    = (int)(idx & 15);
  int    lane = (int)((idx >> 4) & 31);
  size_t rest = idx >> 9;
  int    ch   = (int)(rest % (size_t)chunks);
  int    nt   = (int)(rest / (size_t)chunks);
  int n  = (nt << 4) + (lane & 15);
  int kk = (ch << 5) + ((lane >> 4) << 3) + (j < 8 ? j : j + 8);
  float v = 0.0f;
  if (n < 10)       v = piw [(size_t)n        * 1024 + kk];
  else if (n < 50)  v = sigw[(size_t)(n - 10) * 1024 + kk];
  else if (n < 90)  v = muw [(size_t)(n - 50) * 1024 + kk];
  dst[idx] = (_Float16)v;
}

// ---- fused Conv1d(K=5,pad=2) + bias + LayerNorm(C) + ReLU ------------------
// One block = 16 time rows x all 512 output channels, K-dim = Cin*5 = 2560.
// A panel = 20 consecutive time steps async-staged in LDS (zero-padded edges).
__global__ void conv_block_kernel(const _Float16* __restrict__ xin,  // (B,T,Cin) f16
                                  const _Float16* __restrict__ wp,   // packed B frags
                                  const float* __restrict__ bias,
                                  const float* __restrict__ lng,
                                  const float* __restrict__ lnb,
                                  _Float16* __restrict__ yout) {
  extern __shared__ __align__(16) char smem[];
  _Float16* aStage  = (_Float16*)smem;                   // 20*512 f16 (20 KB)
  float*    outTile = (float*)(smem + 20 * Cc * 2);      // 16*512 f32 (32 KB)

  const int tid  = threadIdx.x;
  const int lane = tid & 31;
  const int wv   = tid >> 5;
  const int blk  = blockIdx.x;
  const int b    = blk >> 6;          // T/16 = 64 tiles per batch row
  const int t0   = (blk & 63) << 4;

  // stage x[b, t0-2 .. t0+17, :] via async-to-LDS, zero pad at sequence edges
  const int SEGS = (20 * Cc) / 8;     // 1280 x 16-byte segments
  for (int seg = tid; seg < SEGS; seg += 256) {
    const int row  = seg >> 6;        // 64 segments per 512-wide row
    const int col8 = (seg & 63) << 3;
    const int tt   = t0 - 2 + row;
    _Float16* dst  = aStage + (seg << 3);
    if (tt >= 0 && tt < Tt) {
      async_copy_b128(dst, xin + ((size_t)b * Tt + tt) * Cc + col8);
    } else {
      v8h z = {};
      *(v8h*)dst = z;
    }
  }
  wait_async0();
  __syncthreads();

  v8f acc[4] = {};
  const int chunks = (Cc * KTAP) / 32;  // 80
  for (int ch = 0; ch < chunks; ++ch) {
    const int kk0  = ch << 5;
    const int ktap = kk0 >> 9;          // kk / 512
    const int cin0 = kk0 & 511;
    // row m of the tile reads time (m + ktap) -> shift base by ktap rows
    v16h a = load_a_frag(aStage + ktap * Cc + cin0, Cc, lane);
#pragma unroll
    for (int i = 0; i < 4; ++i) {
      const int nt = (wv << 2) + i;
      const _Float16* bp = wp + (((size_t)nt * chunks + ch) * 32 + lane) * 16;
      if (ch + 1 < chunks) __builtin_prefetch(bp + 32 * 16, 0, 1);
      v16h bfr = *(const v16h*)bp;
      acc[i] = wmma_f16(a, bfr, acc[i]);
    }
  }

  // accum + bias -> LDS f32 tile
#pragma unroll
  for (int i = 0; i < 4; ++i) {
    const int n0    = ((wv << 2) + i) << 4;
    const int col   = n0 + (lane & 15);
    const int rbase = (lane >> 4) << 3;
    const float bc  = bias[col];
#pragma unroll
    for (int r = 0; r < 8; ++r)
      outTile[(rbase + r) * Cc + col] = acc[i][r] + bc;
  }
  __syncthreads();

  // LayerNorm over channels + ReLU; wave w handles rows 2w, 2w+1 (wave32)
  for (int rr = 0; rr < 2; ++rr) {
    const int row = wv * 2 + rr;
    float s = 0.0f, sq = 0.0f;
    for (int i = 0; i < 16; ++i) {
      float x = outTile[row * Cc + lane + 32 * i];
      s += x; sq += x * x;
    }
    for (int off = 16; off > 0; off >>= 1) {
      s  += __shfl_xor(s,  off, 32);
      sq += __shfl_xor(sq, off, 32);
    }
    const float mean = s * (1.0f / Cc);
    const float var  = sq * (1.0f / Cc) - mean * mean;
    const float rs   = rsqrtf(var + 1e-5f);
    const size_t orow = ((size_t)b * Tt + t0 + row) * Cc;
    for (int i = 0; i < 16; ++i) {
      int col = lane + 32 * i;
      float x = outTile[row * Cc + col];
      float y = (x - mean) * rs * lng[col] + lnb[col];
      yout[orow + col] = (_Float16)fmaxf(y, 0.0f);
    }
  }
}

// ---- xg0 = enc @ W_ih0[:, :512]^T + prenet(prev) @ W_ih0[:, 512:]^T + b ----
// grid (4096 m-tiles, 3 n-groups of 512); output f16 (B,T,1536)
__global__ void xg0_kernel(const _Float16* __restrict__ enc,   // (B,T,512) f16
                           const _Float16* __restrict__ wp,    // packed W_ih0[:, :512]
                           const float* __restrict__ w_ih0,    // (1536, 516) f32
                           const float* __restrict__ b_ih0,
                           const float* __restrict__ prenet_w, // (4,4)
                           const float* __restrict__ prenet_b,
                           const float* __restrict__ target,   // (B,T,4)
                           _Float16* __restrict__ xg0) {
  extern __shared__ __align__(16) char smem[];
  _Float16* aStage = (_Float16*)smem;  // 16*512 f16
  const int tid  = threadIdx.x;
  const int lane = tid & 31;
  const int wv   = tid >> 5;
  const size_t gm0 = (size_t)blockIdx.x * 16;

  for (int seg = tid; seg < (16 * Cc) / 8; seg += 256)
    async_copy_b128(aStage + (seg << 3), enc + gm0 * Cc + (seg << 3));
  wait_async0();
  __syncthreads();

  v8f acc[4] = {};
  const int chunks = Cc / 32;  // 16
  const int ntb = blockIdx.y * 32 + (wv << 2);
  for (int ch = 0; ch < chunks; ++ch) {
    v16h a = load_a_frag(aStage + (ch << 5), Cc, lane);
#pragma unroll
    for (int i = 0; i < 4; ++i) {
      const int nt = ntb + i;
      v16h bfr = *(const v16h*)(wp + (((size_t)nt * chunks + ch) * 32 + lane) * 16);
      acc[i] = wmma_f16(a, bfr, acc[i]);
    }
  }

#pragma unroll
  for (int i = 0; i < 4; ++i) {
    const int col   = ((ntb + i) << 4) + (lane & 15);
    const int rbase = (lane >> 4) << 3;
    const float bc  = b_ih0[col];
    float wtail[DOUT];
#pragma unroll
    for (int d = 0; d < DOUT; ++d) wtail[d] = w_ih0[(size_t)col * 516 + 512 + d];
#pragma unroll
    for (int r = 0; r < 8; ++r) {
      const size_t gm = gm0 + rbase + r;
      const int t = (int)(gm & 1023);
      float prev[DOUT];
#pragma unroll
      for (int e = 0; e < DOUT; ++e)
        prev[e] = (t > 0) ? target[(gm - 1) * DOUT + e] : 0.0f;
      float pc = 0.0f;
#pragma unroll
      for (int d = 0; d < DOUT; ++d) {
        float pn = prenet_b[d];
#pragma unroll
        for (int e = 0; e < DOUT; ++e) pn += prev[e] * prenet_w[d * DOUT + e];
        pc += pn * wtail[d];
      }
      xg0[gm * G3H + col] = (_Float16)(acc[i][r] + bc + pc);
    }
  }
}

// ---- persistent 2-layer GRU scan with zoneout ------------------------------
// 4 blocks, each owns 16 batch rows; h0/h1 live in LDS (f16) for all 1024
// steps. Per step: 3 WMMA GEMMs (16x1536, K=512) + elementwise gate math.
__global__ void gru_kernel(const _Float16* __restrict__ xg0,   // (B,T,1536) f16
                           const _Float16* __restrict__ whh0p,
                           const _Float16* __restrict__ wih1p,
                           const _Float16* __restrict__ whh1p,
                           const float* __restrict__ b_hh0,
                           const float* __restrict__ b_ih1,
                           const float* __restrict__ b_hh1,
                           _Float16* __restrict__ h1seq) {    // (B,T,512) f16
  extern __shared__ __align__(16) char smem[];
  _Float16* h0 = (_Float16*)smem;         // 16*512
  _Float16* h1 = h0 + 16 * Hh;            // 16*512
  _Float16* gA = h1 + 16 * Hh;            // 16*1536
  _Float16* gB = gA + 16 * G3H;           // 16*1536  -> 131072 B total

  const int tid  = threadIdx.x;
  const int lane = tid & 31;
  const int wv   = tid >> 5;
  const int m0   = blockIdx.x * 16;       // batch rows [m0, m0+16)
  const int chunks = Hh / 32;             // 16

  for (int idx = tid; idx < 16 * Hh; idx += 256) {
    h0[idx] = (_Float16)0.0f;
    h1[idx] = (_Float16)0.0f;
  }
  __syncthreads();

  for (int t = 0; t < Tt; ++t) {
    // ---- gB = h0 @ W_hh0^T ----
    {
      v8f acc[12] = {};
      for (int ch = 0; ch < chunks; ++ch) {
        v16h a = load_a_frag(h0 + (ch << 5), Hh, lane);
#pragma unroll
        for (int i = 0; i < 12; ++i) {
          const int nt = wv * 12 + i;
          v16h bfr = *(const v16h*)(whh0p + (((size_t)nt * chunks + ch) * 32 + lane) * 16);
          acc[i] = wmma_f16(a, bfr, acc[i]);
        }
      }
#pragma unroll
      for (int i = 0; i < 12; ++i) {
        const int col   = ((wv * 12 + i) << 4) + (lane & 15);
        const int rbase = (lane >> 4) << 3;
#pragma unroll
        for (int r = 0; r < 8; ++r)
          gB[(rbase + r) * G3H + col] = (_Float16)acc[i][r];
      }
    }
    __syncthreads();

    // ---- h0 gate update (zoneout) ----
    for (int idx = tid; idx < 16 * Hh; idx += 256) {
      const int row = idx >> 9, d = idx & 511;
      const size_t gm = (size_t)(m0 + row) * Tt + t;
      const float xr = (float)xg0[gm * G3H + d];
      const float xz = (float)xg0[gm * G3H + 512 + d];
      const float xn = (float)xg0[gm * G3H + 1024 + d];
      const float hr = (float)gB[row * G3H + d]        + b_hh0[d];
      const float hz = (float)gB[row * G3H + 512 + d]  + b_hh0[512 + d];
      const float hn = (float)gB[row * G3H + 1024 + d] + b_hh0[1024 + d];
      const float r = sigm(xr + hr);
      const float z = sigm(xz + hz);
      const float n = tanhf(xn + r * hn);
      const float hp = (float)h0[idx];
      const float hnew = (1.0f - z) * n + z * hp;
      h0[idx] = (_Float16)(ZONEOUT_F * hp + (1.0f - ZONEOUT_F) * hnew);
    }
    __syncthreads();

    // ---- gA = h0 @ W_ih1^T ; gB = h1 @ W_hh1^T ----
    {
      {
        v8f acc[12] = {};
        for (int ch = 0; ch < chunks; ++ch) {
          v16h a = load_a_frag(h0 + (ch << 5), Hh, lane);
#pragma unroll
          for (int i = 0; i < 12; ++i) {
            const int nt = wv * 12 + i;
            v16h bfr = *(const v16h*)(wih1p + (((size_t)nt * chunks + ch) * 32 + lane) * 16);
            acc[i] = wmma_f16(a, bfr, acc[i]);
          }
        }
#pragma unroll
        for (int i = 0; i < 12; ++i) {
          const int col   = ((wv * 12 + i) << 4) + (lane & 15);
          const int rbase = (lane >> 4) << 3;
#pragma unroll
          for (int r = 0; r < 8; ++r)
            gA[(rbase + r) * G3H + col] = (_Float16)acc[i][r];
        }
      }
      {
        v8f acc[12] = {};
        for (int ch = 0; ch < chunks; ++ch) {
          v16h a = load_a_frag(h1 + (ch << 5), Hh, lane);
#pragma unroll
          for (int i = 0; i < 12; ++i) {
            const int nt = wv * 12 + i;
            v16h bfr = *(const v16h*)(whh1p + (((size_t)nt * chunks + ch) * 32 + lane) * 16);
            acc[i] = wmma_f16(a, bfr, acc[i]);
          }
        }
#pragma unroll
        for (int i = 0; i < 12; ++i) {
          const int col   = ((wv * 12 + i) << 4) + (lane & 15);
          const int rbase = (lane >> 4) << 3;
#pragma unroll
          for (int r = 0; r < 8; ++r)
            gB[(rbase + r) * G3H + col] = (_Float16)acc[i][r];
        }
      }
    }
    __syncthreads();

    // ---- h1 gate update (zoneout) + emit h1_seq ----
    for (int idx = tid; idx < 16 * Hh; idx += 256) {
      const int row = idx >> 9, d = idx & 511;
      const float xr = (float)gA[row * G3H + d]        + b_ih1[d];
      const float xz = (float)gA[row * G3H + 512 + d]  + b_ih1[512 + d];
      const float xn = (float)gA[row * G3H + 1024 + d] + b_ih1[1024 + d];
      const float hr = (float)gB[row * G3H + d]        + b_hh1[d];
      const float hz = (float)gB[row * G3H + 512 + d]  + b_hh1[512 + d];
      const float hn = (float)gB[row * G3H + 1024 + d] + b_hh1[1024 + d];
      const float r = sigm(xr + hr);
      const float z = sigm(xz + hz);
      const float n = tanhf(xn + r * hn);
      const float hp = (float)h1[idx];
      const float hnew = (1.0f - z) * n + z * hp;
      const float hf = ZONEOUT_F * hp + (1.0f - ZONEOUT_F) * hnew;
      h1[idx] = (_Float16)hf;
      h1seq[((size_t)(m0 + row) * Tt + t) * Hh + d] = (_Float16)hf;
    }
    __syncthreads();
  }
}

// ---- heads: hcs=[h1|enc] (K=1024) -> pi(softmax,10) sigma(exp,40) mu(40) ---
constexpr size_t OUT_BASE_PI = (size_t)Bb * Tt * DOUT;                 // 262144
constexpr size_t OUT_BASE_SG = OUT_BASE_PI + (size_t)Bb * Tt * Gg;     // 917504
constexpr size_t OUT_BASE_MU = OUT_BASE_SG + (size_t)Bb * Tt * Gg * DOUT; // 3538944

__global__ void heads_kernel(const _Float16* __restrict__ h1seq,
                             const _Float16* __restrict__ enc,
                             const _Float16* __restrict__ wph,   // packed, Npad=96
                             const float* __restrict__ pib,
                             const float* __restrict__ sigb,
                             const float* __restrict__ mub,
                             float* __restrict__ out) {
  extern __shared__ __align__(16) char smem[];
  _Float16* aStage = (_Float16*)smem;                 // 16*1024 f16 (32 KB)
  float*    ot     = (float*)(smem + 16 * 1024 * 2);  // 16*96 f32

  const int tid  = threadIdx.x;
  const int lane = tid & 31;
  const int wv   = tid >> 5;
  const size_t gm0 = (size_t)blockIdx.x * 16;

  // async-stage hcs rows: [h1 | enc]; 8-half segments never straddle the split
  for (int seg = tid; seg < (16 * 1024) / 8; seg += 256) {
    const int row = seg >> 7;           // 128 segments per 1024-wide row
    const int k8  = (seg & 127) << 3;
    const _Float16* src = (k8 < 512) ? (h1seq + (gm0 + row) * Hh + k8)
                                     : (enc  + (gm0 + row) * Cc + (k8 - 512));
    async_copy_b128(aStage + (seg << 3), src);
  }
  wait_async0();
  __syncthreads();

  if (wv < 6) {
    v8f acc = {};
    for (int ch = 0; ch < 32; ++ch) {
      v16h a   = load_a_frag(aStage + (ch << 5), 1024, lane);
      v16h bfr = *(const v16h*)(wph + (((size_t)wv * 32 + ch) * 32 + lane) * 16);
      acc = wmma_f16(a, bfr, acc);
    }
    const int col = (wv << 4) + (lane & 15);
    float bias = 0.0f;
    if (col < 10)      bias = pib[col];
    else if (col < 50) bias = sigb[col - 10];
    else if (col < 90) bias = mub[col - 50];
    const int rbase = (lane >> 4) << 3;
#pragma unroll
    for (int r = 0; r < 8; ++r)
      ot[(rbase + r) * 96 + col] = acc[r] + bias;
  }
  __syncthreads();

  if (tid < 16) {
    const int row = tid;
    const size_t gm = gm0 + row;
    float mx = -1e30f;
    for (int g = 0; g < Gg; ++g) mx = fmaxf(mx, ot[row * 96 + g]);
    float e[Gg], s = 0.0f;
    for (int g = 0; g < Gg; ++g) { e[g] = __expf(ot[row * 96 + g] - mx); s += e[g]; }
    const float inv = 1.0f / s;
    for (int g = 0; g < Gg; ++g) out[OUT_BASE_PI + gm * Gg + g] = e[g] * inv;
  }
  for (int idx = tid; idx < 16 * 80; idx += 256) {
    const int row = idx / 80, c = idx % 80;
    const size_t gm = gm0 + row;
    if (c < 40) out[OUT_BASE_SG + gm * 40 + c] = __expf(ot[row * 96 + 10 + c]);
    else        out[OUT_BASE_MU + gm * 40 + (c - 40)] = ot[row * 96 + 50 + (c - 40)];
  }
}

// ---------------------------------------------------------------------------
// Host side
// ---------------------------------------------------------------------------
// Workspace layout (bytes):
constexpr size_t SZ_ACT    = (size_t)Bb * Tt * Cc * 2;        // 67,108,864
constexpr size_t SZ_XG0    = (size_t)Bb * Tt * G3H * 2;       // 201,326,592
constexpr size_t SZ_WCONV  = (size_t)(Cc * KTAP) * Cc * 2;    // 2,621,440
constexpr size_t SZ_WLIN   = (size_t)Cc * G3H * 2;            // 1,572,864
constexpr size_t SZ_WHEAD  = (size_t)1024 * 96 * 2;           //   196,608

constexpr size_t OFF_ENCF16 = 0;                         // also reused for h1seq
constexpr size_t OFF_ACT1   = OFF_ENCF16 + SZ_ACT;
constexpr size_t OFF_ACT2   = OFF_ACT1   + SZ_ACT;
constexpr size_t OFF_XG0    = OFF_ACT2   + SZ_ACT;
constexpr size_t OFF_W1P    = OFF_XG0    + SZ_XG0;
constexpr size_t OFF_W2P    = OFF_W1P    + SZ_WCONV;
constexpr size_t OFF_WIH0P  = OFF_W2P    + SZ_WCONV;
constexpr size_t OFF_WHH0P  = OFF_WIH0P  + SZ_WLIN;
constexpr size_t OFF_WIH1P  = OFF_WHH0P  + SZ_WLIN;
constexpr size_t OFF_WHH1P  = OFF_WIH1P  + SZ_WLIN;
constexpr size_t OFF_WHEADP = OFF_WHH1P  + SZ_WLIN;

extern "C" void kernel_launch(void* const* d_in, const int* in_sizes, int n_in,
                              void* d_out, int out_size, void* d_ws, size_t ws_size,
                              hipStream_t stream) {
  const float* enc_in   = (const float*)d_in[0];
  const float* target   = (const float*)d_in[1];
  const float* conv_w1  = (const float*)d_in[2];
  const float* conv_b1  = (const float*)d_in[3];
  const float* ln_g1    = (const float*)d_in[4];
  const float* ln_b1    = (const float*)d_in[5];
  const float* conv_w2  = (const float*)d_in[6];
  const float* conv_b2  = (const float*)d_in[7];
  const float* ln_g2    = (const float*)d_in[8];
  const float* ln_b2    = (const float*)d_in[9];
  const float* prenet_w = (const float*)d_in[10];
  const float* prenet_b = (const float*)d_in[11];
  const float* w_ih0    = (const float*)d_in[12];
  const float* w_hh0    = (const float*)d_in[13];
  const float* b_ih0    = (const float*)d_in[14];
  const float* b_hh0    = (const float*)d_in[15];
  const float* w_ih1    = (const float*)d_in[16];
  const float* w_hh1    = (const float*)d_in[17];
  const float* b_ih1    = (const float*)d_in[18];
  const float* b_hh1    = (const float*)d_in[19];
  const float* pi_w     = (const float*)d_in[20];
  const float* pi_b     = (const float*)d_in[21];
  const float* sigma_w  = (const float*)d_in[22];
  const float* sigma_b  = (const float*)d_in[23];
  const float* mu_w     = (const float*)d_in[24];
  const float* mu_b     = (const float*)d_in[25];
  float* out = (float*)d_out;

  char* ws = (char*)d_ws;
  _Float16* encf16 = (_Float16*)(ws + OFF_ENCF16);
  _Float16* h1seq  = (_Float16*)(ws + OFF_ENCF16);  // alias: enc f16 dead after conv1
  _Float16* act1   = (_Float16*)(ws + OFF_ACT1);
  _Float16* act2   = (_Float16*)(ws + OFF_ACT2);
  _Float16* xg0    = (_Float16*)(ws + OFF_XG0);
  _Float16* w1p    = (_Float16*)(ws + OFF_W1P);
  _Float16* w2p    = (_Float16*)(ws + OFF_W2P);
  _Float16* wih0p  = (_Float16*)(ws + OFF_WIH0P);
  _Float16* whh0p  = (_Float16*)(ws + OFF_WHH0P);
  _Float16* wih1p  = (_Float16*)(ws + OFF_WIH1P);
  _Float16* whh1p  = (_Float16*)(ws + OFF_WHH1P);
  _Float16* wheadp = (_Float16*)(ws + OFF_WHEADP);

  // 1) pack weights to f16 fragment layout
  {
    const size_t tconv = (size_t)(Cc / 16) * ((Cc * KTAP) / 32) * 512;  // 1,310,720
    pack_conv_kernel<<<(unsigned)((tconv + 255) / 256), 256, 0, stream>>>(conv_w1, Cc, w1p, tconv);
    pack_conv_kernel<<<(unsigned)((tconv + 255) / 256), 256, 0, stream>>>(conv_w2, Cc, w2p, tconv);
    const size_t tlin = (size_t)(G3H / 16) * (Cc / 32) * 512;           // 786,432
    pack_linear_kernel<<<(unsigned)((tlin + 255) / 256), 256, 0, stream>>>(w_ih0, 516, G3H, Cc / 32, wih0p, tlin);
    pack_linear_kernel<<<(unsigned)((tlin + 255) / 256), 256, 0, stream>>>(w_hh0, 512, G3H, Cc / 32, whh0p, tlin);
    pack_linear_kernel<<<(unsigned)((tlin + 255) / 256), 256, 0, stream>>>(w_ih1, 512, G3H, Cc / 32, wih1p, tlin);
    pack_linear_kernel<<<(unsigned)((tlin + 255) / 256), 256, 0, stream>>>(w_hh1, 512, G3H, Cc / 32, whh1p, tlin);
    const size_t thead = (size_t)6 * 32 * 512;                          // 98,304
    pack_heads_kernel<<<(unsigned)((thead + 255) / 256), 256, 0, stream>>>(pi_w, sigma_w, mu_w, wheadp, thead);
  }

  // 2) encoder_output -> f16
  {
    const size_t n = (size_t)Bb * Tt * Cc;
    cvt_f16_kernel<<<(unsigned)((n + 255) / 256), 256, 0, stream>>>(enc_in, encf16, n);
  }

  // 3) conv blocks (fused conv+bias+LN+ReLU), f16 activations
  {
    const unsigned mtiles = (Bb * Tt) / 16;            // 4096
    const size_t shmem = 20 * Cc * 2 + 16 * Cc * 4;    // 53,248 B
    conv_block_kernel<<<mtiles, 256, shmem, stream>>>(encf16, w1p, conv_b1, ln_g1, ln_b1, act1);
    conv_block_kernel<<<mtiles, 256, shmem, stream>>>(act1,   w2p, conv_b2, ln_g2, ln_b2, act2);
  }

  // 4) xg0 projection (+ prenet tail + bias), f16 (B,T,1536)
  {
    dim3 grid((Bb * Tt) / 16, 3);
    const size_t shmem = 16 * Cc * 2;                  // 16,384 B
    xg0_kernel<<<grid, 256, shmem, stream>>>(act2, wih0p, w_ih0, b_ih0,
                                             prenet_w, prenet_b, target, xg0);
  }

  // 5) persistent GRU scan: 4 blocks x 16 batch rows, 131 KB dynamic LDS
  {
    const size_t shmem = (size_t)(16 * Hh + 16 * Hh + 16 * G3H + 16 * G3H) * 2; // 131,072
    gru_kernel<<<4, 256, shmem, stream>>>(xg0, whh0p, wih1p, whh1p,
                                          b_hh0, b_ih1, b_hh1, h1seq);
  }

  // 6) MDN heads: GEMM + bias + softmax/exp epilogue
  {
    const unsigned mtiles = (Bb * Tt) / 16;
    const size_t shmem = 16 * 1024 * 2 + 16 * 96 * 4;  // 38,912 B
    heads_kernel<<<mtiles, 256, shmem, stream>>>(h1seq, act2, wheadp,
                                                 pi_b, sigma_b, mu_b, out);
  }

  // 7) outs = target_prosody (teacher-forced pass-through)
  {
    const size_t n = (size_t)Bb * Tt * DOUT;
    copy_outs_kernel<<<(unsigned)((n + 255) / 256), 256, 0, stream>>>(target, out, n);
  }
}
